// SimpleRNN_82832739270762
// MI455X (gfx1250) — compile-verified, matrix operand order checked by reading, not run on hardware
//
#include <hip/hip_runtime.h>
#include <hip/hip_bf16.h>
#include <math.h>
#include <stdint.h>

// Problem dims (fixed by reference)
#define B_ 64
#define T_ 1024
#define I_ 1024
#define H_ 1024

typedef __attribute__((ext_vector_type(16))) __bf16 v16bf;
typedef __attribute__((ext_vector_type(8)))  float  v8f;
typedef __attribute__((ext_vector_type(4)))  unsigned int u32x4;
typedef __attribute__((ext_vector_type(8)))  unsigned int u32x8;

// 16 bf16 values = 32 bytes
union Frag16 {
  v16bf bf;
  unsigned int w[8];
  uint4 q[2];
};

// RNE f32->bf16 (one-time prep only)
__device__ __forceinline__ unsigned short f2bf(float x) {
  unsigned int u = __float_as_uint(x);
  u += 0x7FFFu + ((u >> 16) & 1u);
  return (unsigned short)(u >> 16);
}

// Fast packed f32x2 -> bf16x2 (round-to-nearest, v_add + v_perm): lo -> [15:0]
__device__ __forceinline__ unsigned int pk_bf16(float lo, float hi) {
  unsigned int ul = __float_as_uint(lo) + 0x8000u;
  unsigned int uh = __float_as_uint(hi) + 0x8000u;
  return __builtin_amdgcn_perm(uh, ul, 0x07060302u);
}

// gfx1250 async global->LDS copy (ASYNCcnt). VDST vgpr = wave-relative LDS
// byte address (= low 32 bits of the flat shared pointer).
__device__ __forceinline__ void async_load_b128(unsigned int lds_off, const void* gaddr) {
  asm volatile("global_load_async_to_lds_b128 %0, %1, off"
               :: "v"(lds_off), "v"((unsigned long long)(uintptr_t)gaddr)
               : "memory");
}
__device__ __forceinline__ void wait_async0() {
  asm volatile("s_wait_asynccnt 0x0" ::: "memory");
}

// ---------------------------------------------------------------------------
// Tensor Data Mover: one 2D-tile DMA global->LDS (TENSORcnt).
// Tile: 256 rows x 32 bf16 (64 B) with row stride 1024 elements; LDS gets
// 64 B data + 16 B pad per row (pad_interval = 16 DWORDs, pad_amount = 4
// DWORDs) => 80 B/row, identical to the hand-padded layout consumers use.
// ---------------------------------------------------------------------------
__device__ __forceinline__ void tdm_load_w_tile(unsigned int lds_addr,
                                                const unsigned short* gsrc) {
  const unsigned long long ga = (unsigned long long)(uintptr_t)gsrc;
  const u32x4 g0 = {
      1u,                                   // count=1, user descriptor
      lds_addr,                             // D#.lds_addr (bytes)
      (unsigned int)ga,                     // global_addr[31:0]
      ((unsigned int)(ga >> 32) & 0x01FFFFFFu) | 0x80000000u  // addr[56:32]|type=2
  };
  const u32x8 g1 = {
      (1u << 16) | (1u << 20) | (3u << 22) | (3u << 25),
      //  ^data_size=2B  ^pad_en   ^pad_int=16DW ^pad_amt=4DW ; wg_mask=0
      (1024u & 0xFFFFu) << 16,              // tensor_dim0[15:0]  (=1024)
      ((1024u & 0xFFFFu) << 16),            // dim0[31:16]=0 | tensor_dim1[15:0]=1024
      (32u << 16),                          // dim1[31:16]=0 | tile_dim0=32 elems
      256u,                                 // tile_dim1=256 rows (tile_dim2=0)
      1024u,                                // tensor_dim0_stride[31:0] (elems)
      0u,                                   // stride[47:32]=0, dim1_stride lo=0
      0u
  };
  asm volatile("tensor_load_to_lds %0, %1" :: "s"(g0), "s"(g1) : "memory");
}
__device__ __forceinline__ void wait_tensor0() {
  __builtin_amdgcn_s_wait_tensorcnt(0);
}

__device__ __forceinline__ void pack_afrag(Frag16& af, const float4 cA[4]) {
  af.w[0] = pk_bf16(cA[0].x, cA[0].y); af.w[1] = pk_bf16(cA[0].z, cA[0].w);
  af.w[2] = pk_bf16(cA[1].x, cA[1].y); af.w[3] = pk_bf16(cA[1].z, cA[1].w);
  af.w[4] = pk_bf16(cA[2].x, cA[2].y); af.w[5] = pk_bf16(cA[2].z, cA[2].w);
  af.w[6] = pk_bf16(cA[3].x, cA[3].y); af.w[7] = pk_bf16(cA[3].z, cA[3].w);
}

// ---------------------------------------------------------------------------
// Prep: weights -> bf16 (RNE), fold biases, zero barrier state.
// ---------------------------------------------------------------------------
__global__ __launch_bounds__(256) void prep_kernel(
    const float* __restrict__ W_ih, const float* __restrict__ W_hh,
    const float* __restrict__ b_ih, const float* __restrict__ b_hh,
    unsigned short* __restrict__ wih_bf, unsigned short* __restrict__ whh_bf,
    float* __restrict__ bias, unsigned int* __restrict__ bar) {
  size_t i = (size_t)blockIdx.x * blockDim.x + threadIdx.x;
  const size_t NW = (size_t)H_ * I_;
  if (i < NW) {
    wih_bf[i] = f2bf(W_ih[i]);
    whh_bf[i] = f2bf(W_hh[i]);
  }
  if (i < (size_t)H_) bias[i] = b_ih[i] + b_hh[i];
  if (i < 8) bar[i] = 0u;
}

// ---------------------------------------------------------------------------
// Phase 1: x_proj = input @ W_ih^T + bias.
// Block = 8 waves, tile 128(M) x 256(N). B staged in LDS by the Tensor Data
// Mover (wave 0 issues one 2D DMA per 32-K chunk), double buffered.
// A: direct global f32, manually pipelined, packed to bf16 with v_perm.
// B fragments: explicit 2-deep register pipeline so each ds_load's latency
// hides behind the previous WMMA.
// ---------------------------------------------------------------------------
__global__ __launch_bounds__(256) void xproj_kernel(
    const float* __restrict__ A, const unsigned short* __restrict__ Wbf,
    const float* __restrict__ bias, float* __restrict__ C) {
  __shared__ __align__(128) unsigned char ldsB[2 * 20480];

  const int tid   = threadIdx.x;
  const int lane  = tid & 31;
  const int wave  = tid >> 5;
  const int npass = blockIdx.x & 3;   // 4 passes of 256 columns
  const int mblk  = blockIdx.x >> 2;  // 512 blocks of 128 rows
  const int mrow0 = mblk * 128 + wave * 16;
  const int n0    = npass * 256;
  const int l15   = lane & 15;
  const int kb    = (lane < 16) ? 0 : 8;
  const int row   = mrow0 + l15;

  const unsigned int ldsBase = (unsigned int)(uintptr_t)&ldsB[0];
  const unsigned short* gW0 = Wbf + (size_t)n0 * I_;

  // Preload K-chunk 0 into buffer 0 (one TDM DMA for the whole 256x32 tile)
  if (wave == 0) tdm_load_w_tile(ldsBase, gW0);

  const v8f zero = {0.f, 0.f, 0.f, 0.f, 0.f, 0.f, 0.f, 0.f};
  v8f acc[16];
#pragma unroll
  for (int i = 0; i < 16; ++i) acc[i] = zero;

  const float* arow = A + (size_t)row * I_;
  float4 cA[4];
  {
    const float4* a0 = reinterpret_cast<const float4*>(arow + kb);
    const float4* a1 = reinterpret_cast<const float4*>(arow + kb + 16);
    cA[0] = a0[0]; cA[1] = a0[1]; cA[2] = a1[0]; cA[3] = a1[1];
  }

  if (wave == 0) wait_tensor0();
  __syncthreads();

  for (int kk = 0; kk < I_; kk += 32) {
    const int cur = (kk >> 5) & 1;

    // Kick off TDM for the next B chunk into the other buffer
    if (wave == 0 && kk + 32 < I_) {
      tdm_load_w_tile(ldsBase + (unsigned int)(cur ^ 1) * 20480u, gW0 + (kk + 32));
    }

    // Prefetch next A chunk into registers (overlaps the WMMA body)
    float4 nA[4];
    if (kk + 32 < I_) {
      const float4* a0 = reinterpret_cast<const float4*>(arow + kk + 32 + kb);
      const float4* a1 = reinterpret_cast<const float4*>(arow + kk + 32 + kb + 16);
      nA[0] = a0[0]; nA[1] = a0[1]; nA[2] = a1[0]; nA[3] = a1[1];
    }

    Frag16 af;
    pack_afrag(af, cA);

    const unsigned char* bb = &ldsB[(unsigned int)cur * 20480u] + (unsigned int)l15 * 80u + kb * 2;

    // Software-pipelined B fragments (parity-indexed, no register copies):
    // issue load for tile nt+1 before the WMMA for tile nt.
    Frag16 bf2[2];
    bf2[0].q[0] = *reinterpret_cast<const uint4*>(bb);
    bf2[0].q[1] = *reinterpret_cast<const uint4*>(bb + 32);
#pragma unroll
    for (int nt = 0; nt < 16; ++nt) {
      if (nt + 1 < 16) {
        const unsigned char* cp = bb + (unsigned int)((nt + 1) * 16) * 80u;
        bf2[(nt + 1) & 1].q[0] = *reinterpret_cast<const uint4*>(cp);
        bf2[(nt + 1) & 1].q[1] = *reinterpret_cast<const uint4*>(cp + 32);
      }
      acc[nt] = __builtin_amdgcn_wmma_f32_16x16x32_bf16(
          false, af.bf, false, bf2[nt & 1].bf, (short)0, acc[nt], false, false);
    }

    cA[0] = nA[0]; cA[1] = nA[1]; cA[2] = nA[2]; cA[3] = nA[3];

    if (wave == 0) wait_tensor0();  // next buffer's DMA done
    __syncthreads();                // published to all 8 waves
  }

  // Epilogue: + bias, store. D layout: vgpr r -> M = r + (lane>=16?8:0)
  const int mb = mrow0 + ((lane >= 16) ? 8 : 0);
#pragma unroll
  for (int nt = 0; nt < 16; ++nt) {
    const int ncol = n0 + nt * 16 + l15;
    const float bv = bias[ncol];
#pragma unroll
    for (int r = 0; r < 8; ++r) {
      C[(size_t)(mb + r) * H_ + ncol] = acc[nt][r] + bv;
    }
  }
}

// ---------------------------------------------------------------------------
// Grid-wide generation barrier (persistent-kernel device sync).
// ---------------------------------------------------------------------------
__device__ __forceinline__ void grid_barrier(unsigned int* counter,
                                             unsigned int* gen,
                                             unsigned int nb) {
  __threadfence();
  __syncthreads();
  if (threadIdx.x == 0) {
    unsigned int old = __hip_atomic_load(gen, __ATOMIC_RELAXED, __HIP_MEMORY_SCOPE_AGENT);
    unsigned int arrived =
        __hip_atomic_fetch_add(counter, 1u, __ATOMIC_ACQ_REL, __HIP_MEMORY_SCOPE_AGENT);
    if (arrived == nb - 1u) {
      __hip_atomic_store(counter, 0u, __ATOMIC_RELAXED, __HIP_MEMORY_SCOPE_AGENT);
      __hip_atomic_fetch_add(gen, 1u, __ATOMIC_RELEASE, __HIP_MEMORY_SCOPE_AGENT);
    } else {
      while (__hip_atomic_load(gen, __ATOMIC_ACQUIRE, __HIP_MEMORY_SCOPE_AGENT) == old) {
        __builtin_amdgcn_s_sleep(1);
      }
    }
  }
  __syncthreads();
}

// ---------------------------------------------------------------------------
// Phase 2: persistent scan. 32 blocks; block b owns ALL 64 batch rows and
// 32 columns [b*32, b*32+32). Its W_hh slice (32 x 1024 bf16 = 64 KB) is
// staged into LDS ONCE via async copies (XOR-swizzled 16B chunks ->
// conflict-free reads), then 1024 steps only read h_{t-1} (L2-resident).
// ---------------------------------------------------------------------------
__global__ __launch_bounds__(256) void rnn_scan_kernel(
    const float* __restrict__ h0, const unsigned short* __restrict__ Whh,
    float* __restrict__ out, unsigned int* __restrict__ bar) {
  __shared__ __align__(128) unsigned char ldsW[64 * 1024];

  const int tid   = threadIdx.x;
  const int lane  = tid & 31;
  const int wave  = tid >> 5;
  const int mtile = wave & 3;     // 4 tiles cover B=64
  const int ntl   = wave >> 2;    // 0..1
  const int l15   = lane & 15;
  const int kb    = (lane < 16) ? 0 : 8;
  const int rowm  = mtile * 16 + l15;
  const int colbase = blockIdx.x * 32;
  const int col     = colbase + ntl * 16 + l15;

  // ---- Stage the 32-column W_hh slice into LDS (async, XOR swizzle) ----
  {
    const int c = tid >> 3;  // column 0..31
    const int p = tid & 7;   // 8 threads/column, 16 chunks each
    const unsigned short* g = Whh + (size_t)(colbase + c) * H_ + p * 128;
    const unsigned int base =
        (unsigned int)(uintptr_t)&ldsW[0] + (unsigned int)c * 2048u;
#pragma unroll
    for (int i = 0; i < 16; ++i) {
      const unsigned int j = (unsigned int)(p * 16 + i);  // 16B-chunk index
      async_load_b128(base + ((j ^ (unsigned int)(c & 15)) << 4), g + i * 8);
    }
  }
  wait_async0();
  __syncthreads();

  unsigned int* counter = bar;
  unsigned int* gen     = bar + 1;
  const size_t bstride  = (size_t)T_ * H_;
  const v8f zero = {0.f, 0.f, 0.f, 0.f, 0.f, 0.f, 0.f, 0.f};

  const int ccol = ntl * 16 + l15;  // lane's column within block (0..31)
  const unsigned char* wbase = &ldsW[(unsigned int)ccol * 2048u];
  const unsigned int cx = (unsigned int)(ccol & 15);

#pragma unroll 1
  for (int t = 0; t < T_; ++t) {
    const float* arow = (t == 0)
        ? (h0 + (size_t)rowm * H_)
        : (out + (size_t)rowm * bstride + (size_t)(t - 1) * H_);

    v8f acc = zero;
    float4 cA[4];
    {
      const float4* a0 = reinterpret_cast<const float4*>(arow + kb);
      const float4* a1 = reinterpret_cast<const float4*>(arow + kb + 16);
      cA[0] = a0[0]; cA[1] = a0[1]; cA[2] = a1[0]; cA[3] = a1[1];
    }

#pragma unroll 2
    for (int kk = 0; kk < H_; kk += 32) {
      float4 nA[4];
      if (kk + 32 < H_) {
        const float4* a0 = reinterpret_cast<const float4*>(arow + kk + 32 + kb);
        const float4* a1 = reinterpret_cast<const float4*>(arow + kk + 32 + kb + 16);
        nA[0] = a0[0]; nA[1] = a0[1]; nA[2] = a1[0]; nA[3] = a1[1];
      }
      Frag16 af;
      pack_afrag(af, cA);

      const unsigned int j0 = (unsigned int)(kk >> 3) + (unsigned int)(kb >> 3);
      Frag16 bfr;
      bfr.q[0] = *reinterpret_cast<const uint4*>(wbase + ((j0 ^ cx) << 4));
      bfr.q[1] = *reinterpret_cast<const uint4*>(wbase + (((j0 + 2u) ^ cx) << 4));

      acc = __builtin_amdgcn_wmma_f32_16x16x32_bf16(
          false, af.bf, false, bfr.bf, (short)0, acc, false, false);

      cA[0] = nA[0]; cA[1] = nA[1]; cA[2] = nA[2]; cA[3] = nA[3];
    }

    // h_t = tanh(x_proj + h_{t-1} W_hh^T); in-place into out[:, t, :]
    const int mb = mtile * 16 + ((lane >= 16) ? 8 : 0);
#pragma unroll
    for (int r = 0; r < 8; ++r) {
      const size_t idx = (size_t)(mb + r) * bstride + (size_t)t * H_ + col;
      const float v = tanhf(out[idx] + acc[r]);
      out[idx] = v;
      if (t == T_ - 1) {
        out[(size_t)B_ * T_ * H_ + (size_t)(mb + r) * H_ + col] = v;
      }
    }

    grid_barrier(counter, gen, gridDim.x);
  }
}

// ---------------------------------------------------------------------------
extern "C" void kernel_launch(void* const* d_in, const int* in_sizes, int n_in,
                              void* d_out, int out_size, void* d_ws, size_t ws_size,
                              hipStream_t stream) {
  (void)in_sizes; (void)n_in; (void)out_size; (void)ws_size;
  const float* input  = (const float*)d_in[0];  // (B,T,I) f32
  const float* hidden = (const float*)d_in[1];  // (B,H)   f32
  const float* W_ih   = (const float*)d_in[2];  // (H,I)   f32
  const float* b_ih   = (const float*)d_in[3];  // (H,)    f32
  const float* W_hh   = (const float*)d_in[4];  // (H,H)   f32
  const float* b_hh   = (const float*)d_in[5];  // (H,)    f32
  float* out = (float*)d_out;                    // (B,T,H) ++ (B,H)

  char* ws = (char*)d_ws;
  unsigned short* wih_bf = (unsigned short*)ws;                              // 2 MB
  unsigned short* whh_bf = (unsigned short*)(ws + (size_t)2 * 1024 * 1024);  // 2 MB
  float*          bias   = (float*)(ws + (size_t)4 * 1024 * 1024);           // 4 KB
  unsigned int*   bar    = (unsigned int*)(ws + (size_t)4 * 1024 * 1024 + 4096);

  prep_kernel<<<(H_ * I_ + 255) / 256, 256, 0, stream>>>(
      W_ih, W_hh, b_ih, b_hh, wih_bf, whh_bf, bias, bar);

  xproj_kernel<<<(B_ * T_ / 128) * (H_ / 256), 256, 0, stream>>>(
      input, wih_bf, bias, out);

  rnn_scan_kernel<<<32, 256, 0, stream>>>(hidden, whh_bf, out, bar);
}
